// CausalSelfAttention_27891517621075
// MI455X (gfx1250) — matrix operand order repaired and checked by reading, not compiled, not run on hardware
//
#include <hip/hip_runtime.h>
#include <hip/hip_bf16.h>

typedef __attribute__((ext_vector_type(16))) _Float16 v16h;
typedef __attribute__((ext_vector_type(8)))  float    v8f;

// ---------------------------------------------------------------------------
// CDNA5 async global->LDS copy (ASYNCcnt-tracked), with portable fallback.
// Probe result: __builtin_amdgcn_global_load_async_to_lds_b128 takes
//   (v4i addrspace(1)*, v4i addrspace(3)*, imm offset, imm cpol)
// where v4i = int __attribute__((vector_size(16))).
// ---------------------------------------------------------------------------
#if __has_builtin(__builtin_amdgcn_global_load_async_to_lds_b128) && \
    __has_builtin(__builtin_amdgcn_s_wait_asynccnt)
#define USE_ASYNC_LDS 1
#else
#define USE_ASYNC_LDS 0
#endif

typedef int v4i __attribute__((vector_size(16)));
typedef __attribute__((address_space(1))) v4i v4i_as1;
typedef __attribute__((address_space(3))) v4i v4i_as3;

__device__ __forceinline__ void cp_async_b128(void* lds_dst, const void* gsrc) {
#if USE_ASYNC_LDS
  __builtin_amdgcn_global_load_async_to_lds_b128(
      (v4i_as1*)gsrc, (v4i_as3*)lds_dst, 0, 0);
#else
  *(uint4*)lds_dst = *(const uint4*)gsrc;
#endif
}
__device__ __forceinline__ void cp_async_wait() {
#if USE_ASYNC_LDS
  __builtin_amdgcn_s_wait_asynccnt(0);
#endif
}

// ---------------------------------------------------------------------------
// WMMA wrapper: D = A(16x32 f16) * B(32x16 f16) + C(16x16 f32)
// ---------------------------------------------------------------------------
__device__ __forceinline__ v8f wmma16(v16h a, v16h b, v8f c) {
  return __builtin_amdgcn_wmma_f32_16x16x32_f16(
      false, a, false, b, (short)0, c, false, false);
}

// ---------------------------------------------------------------------------
// Register-layout loaders (this code's contiguous k order, used by A and B alike). Work for global or LDS pointers
// (compiler infers address space -> global_load_* or ds_load_*).
//  A (16x32 f16): lane L -> row r=L&15; lanes 0-15: K=0..15,
//                 lanes 16-31: K=16..31.
//  B (32x16 f16): lane L -> col c=L&15; lanes 0-15: K=0..15, 16-31: K=16..31.
//  C/D (16x16 f32): VGPR v: lanes 0-15 -> M=v, lanes 16-31 -> M=v+8; N=L&15.
// ---------------------------------------------------------------------------
__device__ __forceinline__ v16h load_a_f32(const float* src, int ld) {
  const int l = threadIdx.x & 31, r = l & 15, kb = (l >> 4) << 3;
  v16h a;
#pragma unroll
  for (int e = 0; e < 16; ++e) {
    int k = 2 * kb + e;
    a[e] = (_Float16)src[r * ld + k];
  }
  return a;
}
__device__ __forceinline__ v16h load_a_f16(const _Float16* src, int ld) {
  const int l = threadIdx.x & 31, r = l & 15, kb = (l >> 4) << 3;
  v16h a;
#pragma unroll
  for (int e = 0; e < 16; ++e) {
    int k = 2 * kb + e;
    a[e] = src[r * ld + k];
  }
  return a;
}
__device__ __forceinline__ v16h load_b_f32(const float* src, int ld) {
  const int l = threadIdx.x & 31, c = l & 15, kb = (l >> 4) << 4;
  v16h b;
#pragma unroll
  for (int e = 0; e < 16; ++e) b[e] = (_Float16)src[(kb + e) * ld + c];
  return b;
}
__device__ __forceinline__ v16h load_b_f16(const _Float16* src, int ld) {
  const int l = threadIdx.x & 31, c = l & 15, kb = (l >> 4) << 4;
  v16h b;
#pragma unroll
  for (int e = 0; e < 16; ++e) b[e] = src[(kb + e) * ld + c];
  return b;
}
// B[k][c] = srcT[c][k]   (for S = q * ck^T)
__device__ __forceinline__ v16h load_b_f16_T(const _Float16* src, int ld) {
  const int l = threadIdx.x & 31, c = l & 15, kb = (l >> 4) << 4;
  v16h b;
#pragma unroll
  for (int e = 0; e < 16; ++e) b[e] = src[c * ld + kb + e];
  return b;
}

// ---------------------------------------------------------------------------
// Kernel A: per (b,h) fused compress(conv+LN) -> q-proj -> QK^T -> softmax -> PV
// Rectangular tril(T,Tk) mask means only compressed positions 0..63 are ever
// visible; live k/v and compressed positions >= 64 are mathematically dead.
// ---------------------------------------------------------------------------
__global__ __launch_bounds__(256)
void attn_fused_kernel(const float* __restrict__ x,
                       const float* __restrict__ cached_k,
                       const float* __restrict__ cached_v,
                       const float* __restrict__ w_attn,
                       const float* __restrict__ b_attn,
                       const float* __restrict__ k_conv,
                       const float* __restrict__ v_conv,
                       const float* __restrict__ ln_k_g,
                       const float* __restrict__ ln_k_b,
                       const float* __restrict__ ln_v_g,
                       const float* __restrict__ ln_v_b,
                       _Float16* __restrict__ y_ws) {
  __shared__ _Float16 s_q [64][72];   // q tiles, later reused for P
  __shared__ _Float16 s_ck[64][72];   // compressed K [pos][dim]
  __shared__ _Float16 s_cv[64][72];   // compressed V [pos][dim]
  // Union region (live ranges separated by barriers):
  //   stage1: conv scratch s_sc [64][65] f32
  //   stage2: double-buffered GEMM stages s_x [2][64][36], s_w [2][32][68]
  //   stage3/4: scores s_sc
  __shared__ __align__(16) unsigned char s_un[2 * 64 * 36 * 4 + 2 * 32 * 68 * 4];
  float (*s_sc)[65]    = (float (*)[65])s_un;
  float (*s_x)[64][36] = (float (*)[64][36])s_un;
  float (*s_w)[32][68] = (float (*)[32][68])(s_un + 2 * 64 * 36 * 4);

  const int bh   = blockIdx.x;
  const int b    = bh >> 4;
  const int h    = bh & 15;
  const int tid  = threadIdx.x;
  const int wave = tid >> 5;
  const int lane = tid & 31;

  // ---- Stage 1: depthwise conv (KS=4, stride 4) + LayerNorm, positions 0..63
  if (tid < 64) {
    const int j = tid;
    {
      const float* kc = cached_k + (size_t)bh * (4096 * 64) + (size_t)(4 * j) * 64;
      float sum = 0.f, ssq = 0.f;
      for (int d = 0; d < 64; ++d) {
        float acc = 0.f;
#pragma unroll
      for (int t = 0; t < 4; ++t) acc += kc[t * 64 + d] * k_conv[d * 4 + t];
        s_sc[j][d] = acc; sum += acc; ssq += acc * acc;
      }
      float mu = sum * (1.f / 64.f);
      float rs = rsqrtf(fmaxf(ssq * (1.f / 64.f) - mu * mu, 0.f) + 1e-5f);
      for (int d = 0; d < 64; ++d)
        s_ck[j][d] = (_Float16)((s_sc[j][d] - mu) * rs * ln_k_g[d] + ln_k_b[d]);
    }
    {
      const float* vc = cached_v + (size_t)bh * (4096 * 64) + (size_t)(4 * j) * 64;
      float sum = 0.f, ssq = 0.f;
      float tmp[64];
      for (int d = 0; d < 64; ++d) {
        float acc = 0.f;
#pragma unroll
        for (int t = 0; t < 4; ++t) acc += vc[t * 64 + d] * v_conv[d * 4 + t];
        tmp[d] = acc; sum += acc; ssq += acc * acc;
      }
      float mu = sum * (1.f / 64.f);
      float rs = rsqrtf(fmaxf(ssq * (1.f / 64.f) - mu * mu, 0.f) + 1e-5f);
      for (int d = 0; d < 64; ++d)
        s_cv[j][d] = (_Float16)((tmp[d] - mu) * rs * ln_v_g[d] + ln_v_b[d]);
    }
  }
  __syncthreads();   // conv scratch (s_sc) dead; union region now owned by s_x/s_w

  // ---- Stage 2: q = x[b] @ w_attn[:, h*64 : h*64+64] + b_attn  (WMMA, K=1024)
  // Cooperative async staging: per 32-wide K-chunk copy
  //   x block  64x32 f32 (8 KB)  and  w block 32x64 f32 (8 KB)
  // via GLOBAL_LOAD_ASYNC_TO_LDS_B128, double-buffered to overlap with WMMA.
  const int tr  = wave >> 1;
  const int tc0 = (wave & 1) * 2;
  const int tc1 = tc0 + 1;

  auto issue_stage2 = [&](int buf, int kk) {
    const int n0 = tid * 2;
#pragma unroll
    for (int j = 0; j < 2; ++j) {        // x: 512 B128 transfers, 2/thread
      int n = n0 + j, r = n >> 3, seg = n & 7;
      cp_async_b128(&s_x[buf][r][seg * 4],
                    x + (size_t)(b * 64 + r) * 1024 + kk + seg * 4);
    }
#pragma unroll
    for (int j = 0; j < 2; ++j) {        // w: 512 B128 transfers, 2/thread
      int n = n0 + j, k = n >> 4, seg = n & 15;
      cp_async_b128(&s_w[buf][k][seg * 4],
                    w_attn + (size_t)(kk + k) * 3072 + h * 64 + seg * 4);
    }
  };

  v8f acc0 = {}; v8f acc1 = {};
  issue_stage2(0, 0);
  for (int kk = 0; kk < 1024; kk += 32) {
    const int buf = (kk >> 5) & 1;
    cp_async_wait();
    __syncthreads();
    if (kk + 32 < 1024) issue_stage2(buf ^ 1, kk + 32);
    v16h at  = load_a_f32(&s_x[buf][tr * 16][0], 36);
    v16h bt0 = load_b_f32(&s_w[buf][0][tc0 * 16], 68);
    v16h bt1 = load_b_f32(&s_w[buf][0][tc1 * 16], 68);
    acc0 = wmma16(at, bt0, acc0);
    acc1 = wmma16(at, bt1, acc1);
  }
  {
    const int cidx = lane & 15;
    const int rofs = (lane < 16) ? 0 : 8;
    const float qb0 = b_attn[h * 64 + tc0 * 16 + cidx];
    const float qb1 = b_attn[h * 64 + tc1 * 16 + cidx];
#pragma unroll
    for (int v = 0; v < 8; ++v) {
      int row = tr * 16 + rofs + v;
      s_q[row][tc0 * 16 + cidx] = (_Float16)(acc0[v] + qb0);
      s_q[row][tc1 * 16 + cidx] = (_Float16)(acc1[v] + qb1);
    }
  }
  __syncthreads();   // s_x/s_w dead; union region now owned by scores s_sc

  // ---- Stage 3: S = q @ ck^T * (1/sqrt(64)), causal mask
  {
    v8f sc0 = {}; v8f sc1 = {};
#pragma unroll
    for (int kk = 0; kk < 64; kk += 32) {
      v16h at  = load_a_f16(&s_q[tr * 16][0] + kk, 72);
      v16h bt0 = load_b_f16_T(&s_ck[tc0 * 16][0] + kk, 72);
      v16h bt1 = load_b_f16_T(&s_ck[tc1 * 16][0] + kk, 72);
      sc0 = wmma16(at, bt0, sc0);
      sc1 = wmma16(at, bt1, sc1);
    }
    const int cidx = lane & 15;
    const int rofs = (lane < 16) ? 0 : 8;
#pragma unroll
    for (int v = 0; v < 8; ++v) {
      int row = tr * 16 + rofs + v;
      int c0  = tc0 * 16 + cidx;
      int c1  = tc1 * 16 + cidx;
      s_sc[row][c0] = (c0 <= row) ? sc0[v] * 0.125f : -__builtin_inff();
      s_sc[row][c1] = (c1 <= row) ? sc1[v] * 0.125f : -__builtin_inff();
    }
  }
  __syncthreads();

  // ---- Stage 4: row softmax -> P (f16) into s_q (reuse)
  if (tid < 64) {
    const int row = tid;
    float m = -__builtin_inff();
    for (int c = 0; c <= row; ++c) m = fmaxf(m, s_sc[row][c]);
    float sum = 0.f;
    for (int c = 0; c < 64; ++c) {
      float e = (c <= row) ? __expf(s_sc[row][c] - m) : 0.f;
      s_sc[row][c] = e;
      sum += e;
    }
    const float inv = 1.f / sum;
    for (int c = 0; c < 64; ++c) s_q[row][c] = (_Float16)(s_sc[row][c] * inv);
  }
  __syncthreads();

  // ---- Stage 5: Y = P @ cv ; write y in [B,T,H,hd] layout (f16) to workspace
  {
    v8f y0 = {}; v8f y1 = {};
#pragma unroll
    for (int kk = 0; kk < 64; kk += 32) {
      v16h at  = load_a_f16(&s_q[tr * 16][0] + kk, 72);
      v16h bt0 = load_b_f16(&s_cv[kk][0] + tc0 * 16, 72);
      v16h bt1 = load_b_f16(&s_cv[kk][0] + tc1 * 16, 72);
      y0 = wmma16(at, bt0, y0);
      y1 = wmma16(at, bt1, y1);
    }
    const int cidx = lane & 15;
    const int rofs = (lane < 16) ? 0 : 8;
#pragma unroll
    for (int v = 0; v < 8; ++v) {
      int row = tr * 16 + rofs + v;
      size_t base = ((size_t)(b * 64 + row)) * 1024 + h * 64;
      y_ws[base + tc0 * 16 + cidx] = (_Float16)y0[v];
      y_ws[base + tc1 * 16 + cidx] = (_Float16)y1[v];
    }
  }
}

// ---------------------------------------------------------------------------
// Kernel B: out = Y[512x1024] @ w_proj[1024x1024] + b_proj   (WMMA)
// Y tile (64x32 f16) staged via async DMA to LDS, shared by all 8 waves
// (removes the 8x redundant A loads); w_proj columns are per-wave unique.
// ---------------------------------------------------------------------------
__global__ __launch_bounds__(256)
void proj_kernel(const _Float16* __restrict__ y,
                 const float* __restrict__ w_proj,
                 const float* __restrict__ b_proj,
                 float* __restrict__ out) {
  __shared__ __align__(16) _Float16 s_y[2][64][40];

  const int tid  = threadIdx.x;
  const int br   = blockIdx.x >> 3;
  const int bc   = blockIdx.x & 7;
  const int wave = tid >> 5;
  const int lane = tid & 31;
  const int col0 = bc * 128 + wave * 16;
  const int row0 = br * 64;

  auto issue_y = [&](int buf, int kk) {   // 64x32 f16 = 256 B128, 1/thread
    int r = tid >> 2, seg = tid & 3;
    cp_async_b128(&s_y[buf][r][seg * 8],
                  y + (size_t)(row0 + r) * 1024 + kk + seg * 8);
  };

  v8f acc[4] = {};
  issue_y(0, 0);
  for (int kk = 0; kk < 1024; kk += 32) {
    const int buf = (kk >> 5) & 1;
    cp_async_wait();
    __syncthreads();
    if (kk + 32 < 1024) issue_y(buf ^ 1, kk + 32);
    __builtin_prefetch(w_proj + (size_t)(kk + 32) * 1024 + col0, 0, 3);
    v16h bt = load_b_f32(w_proj + (size_t)kk * 1024 + col0, 1024);
#pragma unroll
    for (int rt = 0; rt < 4; ++rt) {
      v16h at = load_a_f16(&s_y[buf][rt * 16][0], 40);
      acc[rt] = wmma16(at, bt, acc[rt]);
    }
  }

  const int cidx  = lane & 15;
  const int rofs  = (lane < 16) ? 0 : 8;
  const float bias = b_proj[col0 + cidx];
#pragma unroll
  for (int rt = 0; rt < 4; ++rt) {
#pragma unroll
    for (int v = 0; v < 8; ++v) {
      int row = row0 + rt * 16 + rofs + v;
      out[(size_t)row * 1024 + col0 + cidx] = acc[rt][v] + bias;
    }
  }
}

// ---------------------------------------------------------------------------
extern "C" void kernel_launch(void* const* d_in, const int* in_sizes, int n_in,
                              void* d_out, int out_size, void* d_ws, size_t ws_size,
                              hipStream_t stream) {
  (void)in_sizes; (void)n_in; (void)out_size; (void)ws_size;
  const float* x        = (const float*)d_in[0];
  const float* cached_k = (const float*)d_in[1];
  const float* cached_v = (const float*)d_in[2];
  const float* w_attn   = (const float*)d_in[3];
  const float* b_attn   = (const float*)d_in[4];
  const float* w_proj   = (const float*)d_in[5];
  const float* b_proj   = (const float*)d_in[6];
  const float* k_conv   = (const float*)d_in[7];
  const float* v_conv   = (const float*)d_in[8];
  const float* ln_k_g   = (const float*)d_in[9];
  const float* ln_k_b   = (const float*)d_in[10];
  const float* ln_v_g   = (const float*)d_in[11];
  const float* ln_v_b   = (const float*)d_in[12];

  _Float16* y_ws = (_Float16*)d_ws;        // 512*1024 f16 = 1 MB scratch
  float*    out  = (float*)d_out;

  attn_fused_kernel<<<dim3(128), dim3(256), 0, stream>>>(
      x, cached_k, cached_v, w_attn, b_attn, k_conv, v_conv,
      ln_k_g, ln_k_b, ln_v_g, ln_v_b, y_ws);
  proj_kernel<<<dim3(64), dim3(256), 0, stream>>>(y_ws, w_proj, b_proj, out);
}
